// PAConv_88587995447584
// MI455X (gfx1250) — matrix operand order repaired and checked by reading, not compiled, not run on hardware
//
#include <hip/hip_runtime.h>
#include <hip/hip_bf16.h>
#include <cstdint>

// ---------------------------------------------------------------------------
// PAConv pipeline for MI455X (gfx1250, wave32, WMMA + async global->LDS).
// Multi-pass: GEMM(pre-act) -> per-channel stats -> next GEMM fuses bn+relu.
// GEMMs use v_wmma_f32_16x16x32_f16 with operand tiles staged to LDS via
// GLOBAL_LOAD_ASYNC_TO_LDS_B128 (ASYNCcnt) for fully-coalesced row reads.
// Problem is ~9 GFLOP and HBM/L2-bandwidth bound, so f16 operands are a
// bandwidth win, never a throughput limiter.
// ---------------------------------------------------------------------------

typedef __attribute__((ext_vector_type(16))) _Float16 v16h;
typedef __attribute__((ext_vector_type(8)))  float    v8f;

#define EPS_BN 1e-5f
#define Bsz  4
#define Nsz  4096
#define Ksz  32
#define NKsz (Nsz * Ksz)      // 131072 columns per batch for ScoreNet
#define CIN  64
#define COUT 64

// --- wave32 f16 WMMA lane layouts (CDNA5 ISA §7.12.2) -----------
// A 16x32 (MxK): lane -> row M = lane&15 ; half h -> K:
//   lanes 0-15 : V0..V3 hold K=0..7, V4..V7 hold K=16..23
//   lanes 16-31: V0..V3 hold K=8..15, V4..V7 hold K=24..31
__device__ __forceinline__ int a_k_of(int h, int lane) {
  return ((h & 8) << 1) + (h & 7) + (((lane >> 4) & 1) << 3);
}
// B 32x16 (KxN): lane -> col N = lane&15 ; half h -> K same as A (table order)
__device__ __forceinline__ int b_k_of(int h, int lane) {
  return ((h & 8) << 1) + (h & 7) + (((lane >> 4) & 1) << 3);
}
// C/D 16x16 f32: VGPR r, lane l -> (M = r + 8*(l>=16), N = l&15)

__device__ __forceinline__ v8f wmma16x16x32(v16h a, v16h b, v8f c) {
  return __builtin_amdgcn_wmma_f32_16x16x32_f16(
      /*neg_a=*/false, a, /*neg_b=*/false, b,
      /*c_mod=*/(short)0, c, /*reuse_a=*/false, /*reuse_b=*/false);
}

// Async copy of 16 bytes global -> LDS (VGLOBAL, tracked by ASYNCcnt).
// LDS destination = low 32 bits of the generic address (aperture rule).
__device__ __forceinline__ void async_g2l_b128(const float* gptr, void* lptr) {
  unsigned lds = (unsigned)(uintptr_t)lptr;
  unsigned long long ga = (unsigned long long)(uintptr_t)gptr;
  asm volatile("global_load_async_to_lds_b128 %0, %1, off"
               :: "v"(lds), "v"(ga) : "memory");
}
__device__ __forceinline__ void wait_async0() {
  asm volatile("s_wait_asynccnt 0" ::: "memory");
}

// ---------------------------------------------------------------------------
// Unified LDS-staged GEMM over layout (B, CI, L) -> (B, CO, L):
//   out = W(CO x CI) @ [relu(bn(in)) if NORM else in]
// 256 threads / block; block owns a 128-column tile; each of 8 waves owns a
// 16-column subtile. The CI x 128 f32 input tile is staged into LDS with
// async b128 transfers (512B per row per instruction, fully coalesced), then
// waves read WMMA B-fragments from LDS (f32->f16, bn scale/shift fused).
// CI not a multiple of 32 (e.g. 66) is zero-padded in the fragment read.
// ---------------------------------------------------------------------------
template <int CI, int CO, bool NORM, int L>
__global__ void k_gemm_lds(const float* __restrict__ in,
                           const float* __restrict__ ss,
                           const float* __restrict__ w,
                           float* __restrict__ out) {
  constexpr int KC = (CI + 31) / 32;          // K-chunks of 32
  __shared__ float sbuf[CI * 128];

  constexpr int BPB = L / 128;                // blocks per batch
  int  b    = blockIdx.x / BPB;
  long j0   = (long)(blockIdx.x % BPB) * 128;
  int  t    = threadIdx.x;
  int  lane = t & 31;
  int  wv   = t >> 5;                         // 0..7 -> 16-col subtile
  int  col  = lane & 15;
  int  jl   = wv * 16;

  // Stage CI rows x 128 floats: CI*32 float4 transfers across 256 threads.
  for (int q = t; q < CI * 32; q += 256) {
    int row = q >> 5;                         // 32 float4 per row
    int cq  = q & 31;
    async_g2l_b128(in + ((long)(b * CI + row)) * L + j0 + cq * 4,
                   &sbuf[row * 128 + cq * 4]);
  }
  wait_async0();
  __syncthreads();

  // B fragments from LDS (bn+relu fused if NORM), f32 -> f16.
  v16h bt[KC];
  #pragma unroll
  for (int kc = 0; kc < KC; ++kc)
    #pragma unroll
    for (int h = 0; h < 16; ++h) {
      int c = kc * 32 + b_k_of(h, lane);
      float v = (c < CI) ? sbuf[c * 128 + jl + col] : 0.f;
      if constexpr (NORM) v = fmaxf(fmaf(v, ss[c], ss[CI + c]), 0.f);
      bt[kc][h] = (_Float16)v;
    }

  // A fragments from (small, cache-resident) weights; accumulate f32.
  #pragma unroll
  for (int mt = 0; mt < CO / 16; ++mt) {
    v8f acc = {};
    #pragma unroll
    for (int kc = 0; kc < KC; ++kc) {
      v16h at;
      #pragma unroll
      for (int h = 0; h < 16; ++h) {
        int c = kc * 32 + a_k_of(h, lane);
        at[h] = (c < CI) ? (_Float16)w[(mt * 16 + col) * CI + c]
                         : (_Float16)0.f;
      }
      acc = wmma16x16x32(at, bt[kc], acc);
    }
    #pragma unroll
    for (int r = 0; r < 8; ++r) {
      int o = mt * 16 + r + ((lane >> 4) << 3);
      out[((long)(b * CO + o)) * L + j0 + jl + col] = acc[r];
    }
  }
}

// ---------------------------------------------------------------------------
// Per-channel batchnorm stats over layout (B, C, L): one workgroup per
// channel, deterministic tree reduction; emits fused (scale, shift):
//   scale = gamma * rsqrt(var+eps), shift = beta - mean*scale
// ---------------------------------------------------------------------------
__global__ void k_stats(const float* __restrict__ src,
                        const float* __restrict__ gamma,
                        const float* __restrict__ beta,
                        float* __restrict__ ss, int C, int L) {
  __shared__ float s1[256], s2[256];
  int c = blockIdx.x, t = threadIdx.x;
  float sum = 0.f, sq = 0.f;
  for (int b = 0; b < Bsz; ++b) {
    const float* p = src + ((long)(b * C + c)) * L;
    for (int i = t; i < L; i += 256) { float v = p[i]; sum += v; sq += v * v; }
  }
  s1[t] = sum; s2[t] = sq;
  __syncthreads();
  for (int o = 128; o > 0; o >>= 1) {
    if (t < o) { s1[t] += s1[t + o]; s2[t] += s2[t + o]; }
    __syncthreads();
  }
  if (t == 0) {
    float cnt  = (float)((long)Bsz * L);
    float mean = s1[0] / cnt;
    float var  = s2[0] / cnt - mean * mean;
    float scale = gamma[c] * rsqrtf(var + EPS_BN);
    ss[c]     = scale;
    ss[C + c] = beta[c] - mean * scale;
  }
}

// ---------------------------------------------------------------------------
// Stage 2: y(b,n,k') = sum_c relu(bn(hpre))[b,c,n] * matrice2[c][k']
// 4 waves per (b, 16-n tile); each wave owns 8 of the 32 column tiles.
// ---------------------------------------------------------------------------
__global__ void k_gemm_y(const float* __restrict__ hpre,
                         const float* __restrict__ ss1,
                         const float* __restrict__ m2,
                         float* __restrict__ y) {
  int blk  = blockIdx.x;
  int b    = blk >> 8;
  int n0   = (blk & 255) << 4;
  int lane = threadIdx.x & 31;
  int wid  = threadIdx.x >> 5;
  int m    = lane & 15;

  v16h at[2];                       // A = h^T tile (16 n x 64 c), bn+relu fused
  #pragma unroll
  for (int kc = 0; kc < 2; ++kc)
    #pragma unroll
    for (int h = 0; h < 16; ++h) {
      int c = kc * 32 + a_k_of(h, lane);
      float v = hpre[((long)(b * CIN + c)) * Nsz + n0 + m];
      v = fmaxf(fmaf(v, ss1[c], ss1[CIN + c]), 0.f);
      at[kc][h] = (_Float16)v;
    }

  for (int ct = wid; ct < 32; ct += 4) {
    v8f acc = {};
    #pragma unroll
    for (int kc = 0; kc < 2; ++kc) {
      v16h bt;
      #pragma unroll
      for (int h = 0; h < 16; ++h) {
        int c = kc * 32 + b_k_of(h, lane);
        bt[h] = (_Float16)m2[c * 512 + ct * 16 + m];
      }
      acc = wmma16x16x32(at[kc], bt, acc);
    }
    #pragma unroll
    for (int r = 0; r < 8; ++r) {
      int n = n0 + r + ((lane >> 4) << 3);
      y[((long)(b * Nsz + n)) * 512 + ct * 16 + m] = acc[r];
    }
  }
}

// ---------------------------------------------------------------------------
// ScoreNet stage 3 + softmax (tiny 8x16 matvec per column -> VALU).
// score layout: (B, N*K, 8) so the gather kernel reads 8 contiguous floats.
// ---------------------------------------------------------------------------
__global__ void k_score(const float* __restrict__ s2,
                        const float* __restrict__ ss,
                        const float* __restrict__ w3,
                        const float* __restrict__ b3,
                        float* __restrict__ score) {
  long i = (long)blockIdx.x * blockDim.x + threadIdx.x;   // over B*NK
  if (i >= (long)Bsz * NKsz) return;
  int  b = (int)(i / NKsz);
  long j = i - (long)b * NKsz;

  float v[16];
  #pragma unroll
  for (int c = 0; c < 16; ++c) {
    float t = s2[((long)(b * 16 + c)) * NKsz + j];
    v[c] = fmaxf(fmaf(t, ss[c], ss[16 + c]), 0.f);
  }
  float sc[8], mx = -3.0e38f;
  #pragma unroll
  for (int mm = 0; mm < 8; ++mm) {
    float a = b3[mm];
    #pragma unroll
    for (int c = 0; c < 16; ++c) a = fmaf(w3[mm * 16 + c], v[c], a);
    sc[mm] = a; mx = fmaxf(mx, a);
  }
  float s = 0.f;
  #pragma unroll
  for (int mm = 0; mm < 8; ++mm) { sc[mm] = __expf(sc[mm] - mx); s += sc[mm]; }
  float inv = 1.f / s;
  #pragma unroll
  for (int mm = 0; mm < 8; ++mm) score[i * 8 + mm] = sc[mm] * inv;
}

// ---------------------------------------------------------------------------
// Gather + weighted reduce:
//   outpre[b,o,n] = sum_{k,m} score[b,n,k,m] * y[b, idx[b,n,k], m*64+o]
// y (33.5 MB) is L2-resident; 64 o-threads read 256B contiguous per (k,m).
// Output tile staged through padded LDS for coalesced o-major global writes.
// ---------------------------------------------------------------------------
__global__ void k_gather(const float* __restrict__ score,
                         const float* __restrict__ y,
                         const int*   __restrict__ idx,
                         float* __restrict__ outpre) {
  __shared__ float tile[64 * 65];
  int blk = blockIdx.x;             // Bsz * (Nsz/64)
  int b   = blk >> 6;
  int n0  = (blk & 63) << 6;
  int o   = threadIdx.x & 63;
  int ng  = threadIdx.x >> 6;       // 0..3

  for (int it = 0; it < 16; ++it) {
    int nl = it * 4 + ng;
    int n  = n0 + nl;
    const int*   ip = idx + ((long)(b * Nsz + n)) * Ksz;
    const float* sp = score + ((long)b * NKsz + (long)n * Ksz) * 8;
    float acc = 0.f;
    for (int k = 0; k < Ksz; ++k) {
      int np = ip[k];
      const float* yp = y + ((long)(b * Nsz + np)) * 512 + o;
      if (k + 1 < Ksz && o == 0)    // pull next gathered row toward L0/L2
        __builtin_prefetch(y + ((long)(b * Nsz + ip[k + 1])) * 512, 0, 0);
      #pragma unroll
      for (int mm = 0; mm < 8; ++mm)
        acc = fmaf(sp[k * 8 + mm], yp[mm * 64], acc);
    }
    tile[o * 65 + nl] = acc;
  }
  __syncthreads();
  for (int row = ng; row < 64; row += 4)
    outpre[((long)(b * 64 + row)) * Nsz + n0 + o] = tile[row * 65 + o];
}

// ---------------------------------------------------------------------------
// Final bn + relu (elementwise).
// ---------------------------------------------------------------------------
__global__ void k_final(const float* __restrict__ outpre,
                        const float* __restrict__ ss,
                        float* __restrict__ out) {
  long i = (long)blockIdx.x * 256 + threadIdx.x;
  if (i >= (long)Bsz * 64 * Nsz) return;
  int c = (int)((i >> 12) & 63);    // layout (B, 64, N=4096)
  out[i] = fmaxf(fmaf(outpre[i], ss[c], ss[64 + c]), 0.f);
}

// ---------------------------------------------------------------------------
// Host-side launch. Workspace layout (floats), with lifetime-based aliasing:
//   [0]          hpre (1,048,576)         -- reused later as outpre
//   [1,048,576]  y    (8,388,608)
//   [9,437,184]  A: s0pre (33,554,432)    -- reused as s2pre
//   [42,991,616] B: s1pre (16,777,216)    -- reused as score
//   [59,768,832] stats (512)
// Peak ~239 MB.
// ---------------------------------------------------------------------------
extern "C" void kernel_launch(void* const* d_in, const int* in_sizes, int n_in,
                              void* d_out, int out_size, void* d_ws, size_t ws_size,
                              hipStream_t stream) {
  const float* x        = (const float*)d_in[0];
  const float* xyz      = (const float*)d_in[1];
  const float* conv1_w  = (const float*)d_in[2];
  const float* bn1_g    = (const float*)d_in[3];
  const float* bn1_b    = (const float*)d_in[4];
  const float* matrice2 = (const float*)d_in[5];
  const float* sn_w0    = (const float*)d_in[6];
  const float* sn_g0    = (const float*)d_in[7];
  const float* sn_b0    = (const float*)d_in[8];
  const float* sn_w1    = (const float*)d_in[9];
  const float* sn_g1    = (const float*)d_in[10];
  const float* sn_b1    = (const float*)d_in[11];
  const float* sn_w2    = (const float*)d_in[12];
  const float* sn_g2    = (const float*)d_in[13];
  const float* sn_b2    = (const float*)d_in[14];
  const float* sn_w3    = (const float*)d_in[15];
  const float* sn_bias3 = (const float*)d_in[16];
  const float* bn2_g    = (const float*)d_in[17];
  const float* bn2_b    = (const float*)d_in[18];
  const int*   idx      = (const int*)d_in[19];
  float*       out      = (float*)d_out;

  float* ws     = (float*)d_ws;
  float* hpre   = ws;                        // also outpre
  float* y      = ws + 1048576L;
  float* s0pre  = ws + 9437184L;             // also s2pre
  float* s1pre  = ws + 42991616L;            // also score
  float* stats  = ws + 59768832L;
  float* ss1    = stats;                     // 128
  float* ss_s0  = stats + 128;               // 128
  float* ss_s1  = stats + 256;               // 64
  float* ss_s2  = stats + 320;               // 32
  float* ss2    = stats + 384;               // 128
  float* outpre = hpre;                      // alias (hpre dead after k_gemm_y)
  float* s2pre  = s0pre;                     // alias (s0pre dead after stage 1)
  float* score  = s1pre;                     // alias (s1pre dead after stage 2)

  // Stage 1: conv1 + bn1 stats
  k_gemm_lds<64, 64, false, Nsz>
      <<<Bsz * (Nsz / 128), 256, 0, stream>>>(x, nullptr, conv1_w, hpre);
  k_stats<<<64, 256, 0, stream>>>(hpre, bn1_g, bn1_b, ss1, 64, Nsz);
  // Stage 2: y projection (bn+relu fused)
  k_gemm_y<<<Bsz * (Nsz / 16), 128, 0, stream>>>(hpre, ss1, matrice2, y);
  // ScoreNet (66 -> 64 -> 32 -> 16), bn+relu fused into operand loads
  k_gemm_lds<66, 64, false, NKsz>
      <<<Bsz * (NKsz / 128), 256, 0, stream>>>(xyz, nullptr, sn_w0, s0pre);
  k_stats<<<64, 256, 0, stream>>>(s0pre, sn_g0, sn_b0, ss_s0, 64, NKsz);
  k_gemm_lds<64, 32, true, NKsz>
      <<<Bsz * (NKsz / 128), 256, 0, stream>>>(s0pre, ss_s0, sn_w1, s1pre);
  k_stats<<<32, 256, 0, stream>>>(s1pre, sn_g1, sn_b1, ss_s1, 32, NKsz);
  k_gemm_lds<32, 16, true, NKsz>
      <<<Bsz * (NKsz / 128), 256, 0, stream>>>(s1pre, ss_s1, sn_w2, s2pre);
  k_stats<<<16, 256, 0, stream>>>(s2pre, sn_g2, sn_b2, ss_s2, 16, NKsz);
  k_score<<<(Bsz * NKsz) / 256, 256, 0, stream>>>(s2pre, ss_s2, sn_w3, sn_bias3, score);
  // Gather + reduce, bn2 stats, final bn+relu
  k_gather<<<Bsz * (Nsz / 64), 256, 0, stream>>>(score, y, idx, outpre);
  k_stats<<<64, 256, 0, stream>>>(outpre, bn2_g, bn2_b, ss2, 64, Nsz);
  k_final<<<(Bsz * 64 * Nsz) / 256, 256, 0, stream>>>(outpre, ss2, out);
}
